// MultiHeadDotProductAttention_80049600462928
// MI455X (gfx1250) — matrix operand order repaired and checked by reading, not compiled, not run on hardware
//
#include <hip/hip_runtime.h>
#include <hip/hip_bf16.h>

typedef __attribute__((ext_vector_type(16))) __bf16 v16bf;
typedef __attribute__((ext_vector_type(8)))  float  v8f;
typedef __attribute__((ext_vector_type(8)))  __bf16 bf16x8;
typedef __attribute__((ext_vector_type(4)))  int    v4i;

#define Bn 2
#define Sn 2048
#define Dn 1024
#define Hn 16
#define HDn 64
#define BSn (Bn*Sn)   // 4096

#if defined(__has_builtin)
#if __has_builtin(__builtin_amdgcn_global_load_async_to_lds_b128) && \
    __has_builtin(__builtin_amdgcn_s_wait_asynccnt)
#define ASYNC_LDS 1
#endif
#endif

typedef __attribute__((address_space(1))) v4i* gv4i_p;
typedef __attribute__((address_space(3))) v4i* lv4i_p;

__device__ __forceinline__ v8f wmma_bf16(v16bf a, v16bf b, v8f c) {
  return __builtin_amdgcn_wmma_f32_16x16x32_bf16(
      /*neg_a=*/false, a, /*neg_b=*/false, b,
      /*c_mod=*/(short)0, c, /*reuse_a=*/false, /*reuse_b=*/false);
}

// A-matrix 16x32 bf16 fragment (ISA 7.12.2): per-lane K indices form two
// contiguous 8-halfword runs at hf*8 and 16+hf*8. p = row base (16B aligned).
__device__ __forceinline__ v16bf load_a_frag(const __bf16* p, int hf) {
  bf16x8 r0 = *(const bf16x8*)(p + hf * 8);
  bf16x8 r1 = *(const bf16x8*)(p + 16 + hf * 8);
  v16bf f;
#pragma unroll
  for (int i = 0; i < 8; ++i) { f[i] = r0[i]; f[8 + i] = r1[i]; }
  return f;
}

// B-matrix 32x16 bf16 fragment: per-lane K indices form two contiguous
// 8-halfword runs at hf*8 and 16+hf*8. p = this lane's column base (K-major, 16B aligned).
__device__ __forceinline__ v16bf load_b_frag(const __bf16* p, int hf) {
  bf16x8 r0 = *(const bf16x8*)(p + hf * 8);
  bf16x8 r1 = *(const bf16x8*)(p + 16 + hf * 8);
  v16bf f;
#pragma unroll
  for (int i = 0; i < 8; ++i) { f[i] = r0[i]; f[8 + i] = r1[i]; }
  return f;
}

// 16-byte global -> LDS copy: async DMA (no VGPR round-trip) when available.
__device__ __forceinline__ void copy16_g2l(__bf16* lds_dst, const __bf16* gsrc) {
#ifdef ASYNC_LDS
  __builtin_amdgcn_global_load_async_to_lds_b128(
      (gv4i_p)gsrc, (lv4i_p)lds_dst, 0, 0);
#else
  *(bf16x8*)lds_dst = *(const bf16x8*)gsrc;
#endif
}
__device__ __forceinline__ void wait_stage() {
#ifdef ASYNC_LDS
  __builtin_amdgcn_s_wait_asynccnt(0);
#endif
}

// ---------------- fp32 -> bf16 conversion ----------------
__global__ __launch_bounds__(256) void cvt_f32_bf16(const float* __restrict__ in,
                                                    __bf16* __restrict__ out, int n) {
  int i = blockIdx.x * 256 + threadIdx.x;
  if (i < n) out[i] = (__bf16)in[i];
}

// fp32 KxN -> bf16 NxK (transposed) for weight matrices
__global__ __launch_bounds__(256) void cvt_t_f32_bf16(const float* __restrict__ in,
                                                      __bf16* __restrict__ out,
                                                      int K, int N) {
  int i = blockIdx.x * 256 + threadIdx.x;
  if (i < K * N) {
    int k = i / N, n = i - k * N;
    out[(size_t)n * K + k] = (__bf16)in[i];
  }
}

// ---------------- bf16 WMMA GEMM: C = (A*W + bias) * scale ----------------
// A: MxK row-major bf16, WT: NxK row-major bf16 (pre-transposed weights).
// Block tile 64(M) x 128(N), 8 waves, each wave 16x64 (4 WMMAs), K step 32.
// Double-buffered LDS staging (async-to-LDS DMA when the toolchain has it).
// transC: store C transposed (C[n*M+m]) -- used to produce V^T for attention.
__global__ __launch_bounds__(256) void gemm_bf16(
    const __bf16* __restrict__ A, const __bf16* __restrict__ WT,
    const float* __restrict__ bias, float scale, int transC,
    __bf16* __restrict__ Cb, float* __restrict__ Cf,
    int M, int N, int K)
{
  __shared__ __align__(16) __bf16 As[2][64][40];
  __shared__ __align__(16) __bf16 BsT[2][128][40];

  const int tid  = threadIdx.x;
  const int lane = tid & 31;
  const int wave = tid >> 5;
  const int hf   = lane >> 4;
  const int ln   = lane & 15;
  const int wm   = wave & 3;     // M subtile (4 x 16 = 64)
  const int wn   = wave >> 2;    // N group (2 x 64 = 128)
  const int m0   = blockIdx.y * 64;
  const int n0   = blockIdx.x * 128;

  const int arow = tid >> 2;        // 0..63
  const int acol = (tid & 3) * 8;   // 0,8,16,24
  const int brow = tid >> 1;        // 0..127
  const int bcol = (tid & 1) * 16;  // 0,16

  auto stage = [&](int buf, int k0) {
    copy16_g2l(&As[buf][arow][acol],
               &A[(size_t)(m0 + arow) * K + k0 + acol]);
    copy16_g2l(&BsT[buf][brow][bcol],
               &WT[(size_t)(n0 + brow) * K + k0 + bcol]);
    copy16_g2l(&BsT[buf][brow][bcol + 8],
               &WT[(size_t)(n0 + brow) * K + k0 + bcol + 8]);
  };

  v8f acc[4] = {};

  stage(0, 0);
  wait_stage();
  __syncthreads();

  int buf = 0;
  for (int k0 = 0; k0 < K; k0 += 32) {
    if (k0 + 32 < K) stage(buf ^ 1, k0 + 32);  // overlap DMA with WMMAs

    const v16bf af = load_a_frag(&As[buf][wm * 16 + ln][0], hf);
#pragma unroll
    for (int t = 0; t < 4; ++t) {
      const v16bf bfr = load_b_frag(&BsT[buf][wn * 64 + t * 16 + ln][0], hf);
      acc[t] = wmma_bf16(af, bfr, acc[t]);
    }

    wait_stage();     // our async copies for buf^1 have landed
    __syncthreads();  // everyone's copies visible; buf free for reuse
    buf ^= 1;
  }

#pragma unroll
  for (int t = 0; t < 4; ++t) {
    const int n = n0 + wn * 64 + t * 16 + ln;
    const float bv = bias ? bias[n] : 0.0f;
#pragma unroll
    for (int r = 0; r < 8; ++r) {
      const int m = m0 + wm * 16 + r + hf * 8;
      const float val = (acc[t][r] + bv) * scale;
      if (Cb) {
        if (transC) Cb[(size_t)n * M + m] = (__bf16)val;
        else        Cb[(size_t)m * N + n] = (__bf16)val;
      } else {
        Cf[(size_t)m * N + n] = val;
      }
    }
  }
}

// ---------------- causal flash attention ----------------
// Q,K: bf16 [B*S][H*HD] row-major (Q pre-scaled by 1/sqrt(HD)).
// Vt:  bf16 [H*HD][B*S]  (V transposed, so P*V B-fragments are contiguous).
// O:   bf16 [B*S][H*HD].
// One wave per 16-row q-tile; kv streamed in 32-wide tiles.
__global__ __launch_bounds__(256) void flash_attn(
    const __bf16* __restrict__ Q, const __bf16* __restrict__ Km,
    const __bf16* __restrict__ Vt, __bf16* __restrict__ O)
{
  __shared__ __align__(16) __bf16 P[8][16][40];

  const int lane = threadIdx.x & 31;
  const int wave = threadIdx.x >> 5;
  const int hf   = lane >> 4;
  const int ln   = lane & 15;

  const int gw = blockIdx.x * 8 + wave;   // 0..4095
  const int qt = gw & 127;                // q tile within (b,h)
  const int h  = (gw >> 7) & 15;
  const int b  = gw >> 11;

  const size_t base   = ((size_t)b * Sn) * Dn + (size_t)h * HDn;      // Q/K/O
  const size_t vtbase = ((size_t)h * HDn) * BSn + (size_t)b * Sn;     // Vt

  // Q tile 16x64 -> two 16x32 A fragments (contiguous b128 runs)
  v16bf qf[2];
#pragma unroll
  for (int kk = 0; kk < 2; ++kk)
    qf[kk] = load_a_frag(&Q[base + (size_t)(qt * 16 + ln) * Dn + kk * 32], hf);

  float m_i[8], l_i[8], alpha[8];
#pragma unroll
  for (int r = 0; r < 8; ++r) { m_i[r] = -3.0e38f; l_i[r] = 0.0f; }
  v8f o[4] = {};

  const int qmax = qt * 16 + 15;
  for (int j = 0; j * 32 <= qmax; ++j) {
    // scores: Q (16x64) * K^T (64x32) -> two 16x16 tiles.
    // K^T B-fragment: lane's column = kv row, K-dim = d (contiguous in Km).
    v8f sacc[2] = {};
#pragma unroll
    for (int c = 0; c < 2; ++c) {
      const int kvrow = j * 32 + c * 16 + ln;
#pragma unroll
      for (int kk = 0; kk < 2; ++kk) {
        const v16bf kf =
            load_b_frag(&Km[base + (size_t)kvrow * Dn + kk * 32], hf);
        sacc[c] = wmma_bf16(qf[kk], kf, sacc[c]);
      }
    }

    // online softmax (row reductions across 16-lane groups)
    const int kv0 = j * 32 + ln;
    const int kv1 = j * 32 + 16 + ln;
#pragma unroll
    for (int r = 0; r < 8; ++r) {
      const int qg = qt * 16 + r + hf * 8;
      float s0 = (kv0 <= qg) ? sacc[0][r] : -3.0e38f;
      float s1 = (kv1 <= qg) ? sacc[1][r] : -3.0e38f;
      float mx = fmaxf(s0, s1);
#pragma unroll
      for (int off = 1; off < 16; off <<= 1)
        mx = fmaxf(mx, __shfl_xor(mx, off, 16));
      const float mnew = fmaxf(m_i[r], mx);
      alpha[r] = __expf(m_i[r] - mnew);
      const float p0 = __expf(s0 - mnew);
      const float p1 = __expf(s1 - mnew);
      float rs = p0 + p1;
#pragma unroll
      for (int off = 1; off < 16; off <<= 1)
        rs += __shfl_xor(rs, off, 16);
      l_i[r] = l_i[r] * alpha[r] + rs;
      m_i[r] = mnew;
      P[wave][r + hf * 8][ln]      = (__bf16)p0;
      P[wave][r + hf * 8][16 + ln] = (__bf16)p1;
    }

    // rescale running output
#pragma unroll
    for (int t = 0; t < 4; ++t)
#pragma unroll
      for (int r = 0; r < 8; ++r)
        o[t][r] *= alpha[r];

    // P (16x32) * V (32x64): A-frag from LDS (b128), B-frag from Vt (b128).
    const v16bf pf = load_a_frag(&P[wave][ln][0], hf);
#pragma unroll
    for (int t = 0; t < 4; ++t) {
      const v16bf vf = load_b_frag(
          &Vt[vtbase + (size_t)(t * 16 + ln) * BSn + j * 32], hf);
      o[t] = wmma_bf16(pf, vf, o[t]);
    }
  }

  // normalize and store
#pragma unroll
  for (int r = 0; r < 8; ++r) {
    const float inv = 1.0f / l_i[r];
#pragma unroll
    for (int t = 0; t < 4; ++t)
      O[base + (size_t)(qt * 16 + r + hf * 8) * Dn + t * 16 + ln] =
          (__bf16)(o[t][r] * inv);
  }
}

extern "C" void kernel_launch(void* const* d_in, const int* in_sizes, int n_in,
                              void* d_out, int out_size, void* d_ws, size_t ws_size,
                              hipStream_t stream) {
  const float* x  = (const float*)d_in[0];
  // d_in[1] = causal mask (known analytically; unused)
  const float* Wq = (const float*)d_in[2];
  const float* bq = (const float*)d_in[3];
  const float* Wk = (const float*)d_in[4];
  const float* bk = (const float*)d_in[5];
  const float* Wv = (const float*)d_in[6];
  const float* bv = (const float*)d_in[7];
  const float* Wo = (const float*)d_in[8];
  const float* bo = (const float*)d_in[9];

  char* ws = (char*)d_ws;
  const size_t MB = (size_t)1 << 20;
  __bf16* Xb  = (__bf16*)(ws + 0 * MB);   // 8 MB  [B*S][D]
  __bf16* WqT = (__bf16*)(ws + 8 * MB);   // 2 MB each, [N][K]
  __bf16* WkT = (__bf16*)(ws + 10 * MB);
  __bf16* WvT = (__bf16*)(ws + 12 * MB);
  __bf16* WoT = (__bf16*)(ws + 14 * MB);
  __bf16* Qb  = (__bf16*)(ws + 16 * MB);  // 8 MB  [B*S][D]
  __bf16* Kb  = (__bf16*)(ws + 24 * MB);  // 8 MB  [B*S][D]
  __bf16* Vt  = (__bf16*)(ws + 32 * MB);  // 8 MB  [D][B*S]  (transposed)
  __bf16* Ob  = (__bf16*)(ws + 40 * MB);  // 8 MB  [B*S][D]

  const int NX = BSn * Dn;        // 4096*1024
  const int NW = Dn * Hn * HDn;   // 1024*1024

  cvt_f32_bf16<<<NX / 256, 256, 0, stream>>>(x, Xb, NX);
  cvt_t_f32_bf16<<<NW / 256, 256, 0, stream>>>(Wq, WqT, Dn, Dn);
  cvt_t_f32_bf16<<<NW / 256, 256, 0, stream>>>(Wk, WkT, Dn, Dn);
  cvt_t_f32_bf16<<<NW / 256, 256, 0, stream>>>(Wv, WvT, Dn, Dn);
  cvt_t_f32_bf16<<<NW / 256, 256, 0, stream>>>(Wo, WoT, Dn, Dn);

  dim3 gg(Dn / 128, BSn / 64);  // (8, 64)
  // Q scaled by 1/sqrt(HD) AFTER bias, matching reference order
  gemm_bf16<<<gg, 256, 0, stream>>>(Xb, WqT, bq, 0.125f, 0, Qb, nullptr, BSn, Dn, Dn);
  gemm_bf16<<<gg, 256, 0, stream>>>(Xb, WkT, bk, 1.0f,   0, Kb, nullptr, BSn, Dn, Dn);
  gemm_bf16<<<gg, 256, 0, stream>>>(Xb, WvT, bv, 1.0f,   1, Vt, nullptr, BSn, Dn, Dn);

  // 4096 q-tile waves / 8 waves per block
  flash_attn<<<512, 256, 0, stream>>>(Qb, Kb, Vt, Ob);

  gemm_bf16<<<gg, 256, 0, stream>>>(Ob, WoT, bo, 1.0f, 0, nullptr, (float*)d_out,
                                    BSn, Dn, Dn);
}
